// EnhancedSpatialTemporalBlock_9998683865818
// MI455X (gfx1250) — matrix operand order repaired and checked by reading, not compiled, not run on hardware
//
#include <hip/hip_runtime.h>
#include <hip/hip_bf16.h>
#include <math.h>

// ---------------------------------------------------------------------------
// MI455X (gfx1250) implementation. wave32. All matmuls via
// v_wmma_f32_16x16x32_bf16 (f32 accumulate), staged through LDS as bf16.
// QKV activations are kept bf16 in HBM so attention staging can use the
// gfx1250 async global->LDS path (ASYNCcnt) when the builtin is available.
// ---------------------------------------------------------------------------

typedef __attribute__((ext_vector_type(16))) __bf16 bf16x16;
typedef __attribute__((ext_vector_type(8)))  __bf16 bf16x8;
typedef __attribute__((ext_vector_type(8)))  float  f32x8;

#define DEV static __device__ __forceinline__
#define SHUF16(lo, hi) __builtin_shufflevector((lo), (hi), 0,1,2,3,4,5,6,7,8,9,10,11,12,13,14,15)

DEV float gelu_f(float x)    { return 0.5f * x * (1.0f + erff(x * 0.7071067811865475f)); }
DEV float sigmoid_f(float x) { return 1.0f / (1.0f + expf(-x)); }

DEV f32x8 wmma_bf16(bf16x16 a, bf16x16 b, f32x8 c) {
  return __builtin_amdgcn_wmma_f32_16x16x32_bf16(false, a, false, b, (short)0, c, false, false);
}

// ---- gfx1250 async global->LDS copy (16B per lane), guarded by probe ------
#if __has_builtin(__builtin_amdgcn_global_load_async_to_lds_b128)
#define HAVE_ASYNC 1
typedef int v4i_vs __attribute__((vector_size(16)));
typedef __attribute__((address_space(1))) v4i_vs* g_v4i_ptr;
typedef __attribute__((address_space(3))) v4i_vs* l_v4i_ptr;
DEV void async_copy16(void* lds, const void* g) {
  __builtin_amdgcn_global_load_async_to_lds_b128((g_v4i_ptr)(g), (l_v4i_ptr)(lds), 0, 0);
}
DEV void wait_async0() {
#if __has_builtin(__builtin_amdgcn_s_wait_asynccnt)
  __builtin_amdgcn_s_wait_asynccnt(0);
#else
  asm volatile("s_wait_asynccnt 0" ::: "memory");
#endif
}
#else
#define HAVE_ASYNC 0
#endif

// ---------------------------------------------------------------------------
// Generic GEMM:  C[M,N] = act( A[M,K] @ W[N,K]^T + bias )   (optionally += C)
// A: f32 row-major (lda), W: f32 row-major (ldw).  M%64==0, N%64==0, K%32==0.
// Block: 256 threads = 8 waves; tile 64x64; per wave: 16(M) x 32(N).
// ---------------------------------------------------------------------------
template <int ACT, bool ACCUM>
__global__ __launch_bounds__(256) void k_gemm(
    const float* __restrict__ A, int lda,
    const float* __restrict__ W, int ldw,
    const float* __restrict__ bias,
    float* __restrict__ C, int M, int N, int K)
{
  __shared__ __align__(16) __bf16 As[64 * 40];
  __shared__ __align__(16) __bf16 Bs[64 * 40];

  const int tid  = threadIdx.x;
  const int lane = tid & 31;
  const int wv   = tid >> 5;
  const int wm   = wv & 3;
  const int wn   = wv >> 2;
  const int gm   = blockIdx.y * 64;
  const int gn   = blockIdx.x * 64;

  const int lrow = (tid * 8) >> 5;
  const int lcol = (tid * 8) & 31;
  const int l15  = lane & 15;
  const int sel  = lane >> 4;

  f32x8 acc0 = {};
  f32x8 acc1 = {};

  for (int kt = 0; kt < K; kt += 32) {
    {
      const float* pa = A + (size_t)(gm + lrow) * lda + kt + lcol;
      float4 a0 = *(const float4*)(pa);
      float4 a1 = *(const float4*)(pa + 4);
      bf16x8 ha;
      ha[0] = (__bf16)a0.x; ha[1] = (__bf16)a0.y; ha[2] = (__bf16)a0.z; ha[3] = (__bf16)a0.w;
      ha[4] = (__bf16)a1.x; ha[5] = (__bf16)a1.y; ha[6] = (__bf16)a1.z; ha[7] = (__bf16)a1.w;
      *(bf16x8*)&As[lrow * 40 + lcol] = ha;

      const float* pw = W + (size_t)(gn + lrow) * ldw + kt + lcol;
      float4 b0 = *(const float4*)(pw);
      float4 b1 = *(const float4*)(pw + 4);
      bf16x8 hb;
      hb[0] = (__bf16)b0.x; hb[1] = (__bf16)b0.y; hb[2] = (__bf16)b0.z; hb[3] = (__bf16)b0.w;
      hb[4] = (__bf16)b1.x; hb[5] = (__bf16)b1.y; hb[6] = (__bf16)b1.z; hb[7] = (__bf16)b1.w;
      *(bf16x8*)&Bs[lrow * 40 + lcol] = hb;
    }
    __syncthreads();

    const __bf16* pa = &As[(wm * 16 + l15) * 40 + sel * 16];
    bf16x16 afrag = SHUF16(*(const bf16x8*)(pa), *(const bf16x8*)(pa + 8));

    const __bf16* pb0 = &Bs[(wn * 32 + l15) * 40 + sel * 16];
    bf16x16 bfrag0 = SHUF16(*(const bf16x8*)(pb0), *(const bf16x8*)(pb0 + 8));
    const __bf16* pb1 = &Bs[(wn * 32 + 16 + l15) * 40 + sel * 16];
    bf16x16 bfrag1 = SHUF16(*(const bf16x8*)(pb1), *(const bf16x8*)(pb1 + 8));

    acc0 = wmma_bf16(afrag, bfrag0, acc0);
    acc1 = wmma_bf16(afrag, bfrag1, acc1);
    __syncthreads();
  }

#pragma unroll
  for (int r = 0; r < 8; ++r) {
    int m = gm + wm * 16 + sel * 8 + r;
    {
      int n = gn + wn * 32 + l15;
      float v = acc0[r] + (bias ? bias[n] : 0.0f);
      if (ACCUM) v += C[(size_t)m * N + n];
      if (ACT == 1) v = gelu_f(v);
      C[(size_t)m * N + n] = v;
    }
    {
      int n = gn + wn * 32 + 16 + l15;
      float v = acc1[r] + (bias ? bias[n] : 0.0f);
      if (ACCUM) v += C[(size_t)m * N + n];
      if (ACT == 1) v = gelu_f(v);
      C[(size_t)m * N + n] = v;
    }
  }
}

// Same GEMM but writing bf16 output (used for QKV projections).
__global__ __launch_bounds__(256) void k_gemm_obf16(
    const float* __restrict__ A, int lda,
    const float* __restrict__ W, int ldw,
    const float* __restrict__ bias,
    __bf16* __restrict__ C, int M, int N, int K)
{
  __shared__ __align__(16) __bf16 As[64 * 40];
  __shared__ __align__(16) __bf16 Bs[64 * 40];

  const int tid  = threadIdx.x;
  const int lane = tid & 31;
  const int wv   = tid >> 5;
  const int wm   = wv & 3;
  const int wn   = wv >> 2;
  const int gm   = blockIdx.y * 64;
  const int gn   = blockIdx.x * 64;

  const int lrow = (tid * 8) >> 5;
  const int lcol = (tid * 8) & 31;
  const int l15  = lane & 15;
  const int sel  = lane >> 4;

  f32x8 acc0 = {};
  f32x8 acc1 = {};

  for (int kt = 0; kt < K; kt += 32) {
    {
      const float* pa = A + (size_t)(gm + lrow) * lda + kt + lcol;
      float4 a0 = *(const float4*)(pa);
      float4 a1 = *(const float4*)(pa + 4);
      bf16x8 ha;
      ha[0] = (__bf16)a0.x; ha[1] = (__bf16)a0.y; ha[2] = (__bf16)a0.z; ha[3] = (__bf16)a0.w;
      ha[4] = (__bf16)a1.x; ha[5] = (__bf16)a1.y; ha[6] = (__bf16)a1.z; ha[7] = (__bf16)a1.w;
      *(bf16x8*)&As[lrow * 40 + lcol] = ha;

      const float* pw = W + (size_t)(gn + lrow) * ldw + kt + lcol;
      float4 b0 = *(const float4*)(pw);
      float4 b1 = *(const float4*)(pw + 4);
      bf16x8 hb;
      hb[0] = (__bf16)b0.x; hb[1] = (__bf16)b0.y; hb[2] = (__bf16)b0.z; hb[3] = (__bf16)b0.w;
      hb[4] = (__bf16)b1.x; hb[5] = (__bf16)b1.y; hb[6] = (__bf16)b1.z; hb[7] = (__bf16)b1.w;
      *(bf16x8*)&Bs[lrow * 40 + lcol] = hb;
    }
    __syncthreads();

    const __bf16* pa = &As[(wm * 16 + l15) * 40 + sel * 16];
    bf16x16 afrag = SHUF16(*(const bf16x8*)(pa), *(const bf16x8*)(pa + 8));

    const __bf16* pb0 = &Bs[(wn * 32 + l15) * 40 + sel * 16];
    bf16x16 bfrag0 = SHUF16(*(const bf16x8*)(pb0), *(const bf16x8*)(pb0 + 8));
    const __bf16* pb1 = &Bs[(wn * 32 + 16 + l15) * 40 + sel * 16];
    bf16x16 bfrag1 = SHUF16(*(const bf16x8*)(pb1), *(const bf16x8*)(pb1 + 8));

    acc0 = wmma_bf16(afrag, bfrag0, acc0);
    acc1 = wmma_bf16(afrag, bfrag1, acc1);
    __syncthreads();
  }

#pragma unroll
  for (int r = 0; r < 8; ++r) {
    int m = gm + wm * 16 + sel * 8 + r;
    {
      int n = gn + wn * 32 + l15;
      C[(size_t)m * N + n] = (__bf16)(acc0[r] + bias[n]);
    }
    {
      int n = gn + wn * 32 + 16 + l15;
      C[(size_t)m * N + n] = (__bf16)(acc1[r] + bias[n]);
    }
  }
}

// ---------------------------------------------------------------------------
// Fused batched attention. qkv: bf16 (nseq*L, 768) rows = [Q|K|V] per token.
// Grid: (nseq, NH). Block: 2*L threads (= L/16 waves).
// L=64: spatial (with (64,64) f32 additive bias). L=128: temporal (no bias).
// out: f32 (nseq*L, 256), head h occupies cols [h*32, h*32+32).
// ---------------------------------------------------------------------------
template <int L, bool HASBIAS>
__global__ __launch_bounds__(2 * L) void k_attn(
    const __bf16* __restrict__ qkv,
    const float* __restrict__ bias,
    float* __restrict__ out)
{
  constexpr int QSTR = 40;      // Q/K row stride (halfs)
  constexpr int VSTR = 32;      // V row stride (halfs, row-major: [token][hd])
  constexpr int SSTR = L + 8;   // S row stride (halfs)
  __shared__ __align__(16) __bf16 Qs[L * QSTR];
  __shared__ __align__(16) __bf16 Ks[L * QSTR];
  __shared__ __align__(16) __bf16 Vs[L * VSTR];
  __shared__ __align__(16) __bf16 Ss[L * SSTR];

  const int seq = blockIdx.x;
  const int h   = blockIdx.y;
  const int tid = threadIdx.x;

  // stage Q, K, V (bf16 16-byte chunks; async path when available)
#if HAVE_ASYNC
  for (int idx = tid; idx < L * 4; idx += 2 * L) {
    int tok = idx >> 2, ch = idx & 3;
    const __bf16* base = qkv + ((size_t)(seq * L + tok)) * 768 + h * 32 + ch * 8;
    async_copy16(&Qs[tok * QSTR + ch * 8], base);
    async_copy16(&Ks[tok * QSTR + ch * 8], base + 256);
    async_copy16(&Vs[tok * VSTR + ch * 8], base + 512);
  }
  wait_async0();
#else
  for (int idx = tid; idx < L * 4; idx += 2 * L) {
    int tok = idx >> 2, ch = idx & 3;
    const __bf16* base = qkv + ((size_t)(seq * L + tok)) * 768 + h * 32 + ch * 8;
    *(bf16x8*)&Qs[tok * QSTR + ch * 8] = *(const bf16x8*)(base);
    *(bf16x8*)&Ks[tok * QSTR + ch * 8] = *(const bf16x8*)(base + 256);
    *(bf16x8*)&Vs[tok * VSTR + ch * 8] = *(const bf16x8*)(base + 512);
  }
#endif
  __syncthreads();

  const int lane = tid & 31;
  const int wv   = tid >> 5;     // mt tile index, 0..L/16-1
  const int sel  = lane >> 4;
  const int l15  = lane & 15;
  const float scale = 0.17677669529663687f;  // 1/sqrt(32)

  // ---- S = (Q @ K^T) * scale (+bias) ----
  const __bf16* paq = &Qs[(wv * 16 + l15) * QSTR + sel * 16];
  bf16x16 afrag = SHUF16(*(const bf16x8*)(paq), *(const bf16x8*)(paq + 8));
  for (int nt = 0; nt < L / 16; ++nt) {
    const __bf16* pbk = &Ks[(nt * 16 + l15) * QSTR + sel * 16];
    bf16x16 bfrag = SHUF16(*(const bf16x8*)(pbk), *(const bf16x8*)(pbk + 8));
    f32x8 acc = {};
    acc = wmma_bf16(afrag, bfrag, acc);
#pragma unroll
    for (int r = 0; r < 8; ++r) {
      int q  = wv * 16 + sel * 8 + r;
      int ky = nt * 16 + l15;
      float v = acc[r] * scale;
      if (HASBIAS) v += bias[q * 64 + ky];
      Ss[q * SSTR + ky] = (__bf16)v;
    }
  }
  __syncthreads();

  // ---- row softmax (f32 math, bf16 storage) ----
  if (tid < L) {
    float mx = -1e30f;
    for (int j = 0; j < L; ++j) { float v = (float)Ss[tid * SSTR + j]; mx = v > mx ? v : mx; }
    float sum = 0.0f;
    for (int j = 0; j < L; ++j) {
      float e = expf((float)Ss[tid * SSTR + j] - mx);
      sum += e;
      Ss[tid * SSTR + j] = (__bf16)e;
    }
    float inv = 1.0f / sum;
    for (int j = 0; j < L; ++j)
      Ss[tid * SSTR + j] = (__bf16)((float)Ss[tid * SSTR + j] * inv);
  }
  __syncthreads();

  // ---- O = P @ V ----
  for (int nt = 0; nt < 2; ++nt) {
    f32x8 acc = {};
    for (int kk = 0; kk < L / 32; ++kk) {
      const __bf16* pap = &Ss[(wv * 16 + l15) * SSTR + kk * 32 + sel * 16];
      bf16x16 a = SHUF16(*(const bf16x8*)(pap), *(const bf16x8*)(pap + 8));
      bf16x16 b;
#pragma unroll
      for (int j = 0; j < 16; ++j)
        b[j] = Vs[(kk * 32 + sel * 16 + j) * VSTR + nt * 16 + l15];
      acc = wmma_bf16(a, b, acc);
    }
#pragma unroll
    for (int r = 0; r < 8; ++r) {
      int tok = wv * 16 + sel * 8 + r;
      int n   = nt * 16 + l15;
      out[((size_t)(seq * L + tok)) * 256 + h * 32 + n] = acc[r];
    }
  }
}

// ---------------------------------------------------------------------------
// LayerNorm over last dim (256). One block per row, 256 threads.
// out = LN(in + (res ? res : 0)) * w + b
// ---------------------------------------------------------------------------
__global__ __launch_bounds__(256) void k_ln(
    const float* __restrict__ in, const float* __restrict__ res,
    float* __restrict__ out, const float* __restrict__ w, const float* __restrict__ b)
{
  __shared__ float s1[256], s2[256];
  const int row = blockIdx.x, t = threadIdx.x;
  float v = in[(size_t)row * 256 + t];
  if (res) v += res[(size_t)row * 256 + t];
  s1[t] = v; s2[t] = v * v;
  __syncthreads();
  for (int o = 128; o > 0; o >>= 1) {
    if (t < o) { s1[t] += s1[t + o]; s2[t] += s2[t + o]; }
    __syncthreads();
  }
  float m    = s1[0] * (1.0f / 256.0f);
  float var  = s2[0] * (1.0f / 256.0f) - m * m;
  float rinv = rsqrtf(var + 1e-5f);
  out[(size_t)row * 256 + t] = (v - m) * rinv * w[t] + b[t];
}

// conv branch LN + residual: out(b,t,c) = x2(b,t,c) + 0.5 * LN(co(b,c,t))
__global__ __launch_bounds__(256) void k_conv_ln_res(
    const float* __restrict__ co, const float* __restrict__ x2,
    float* __restrict__ out, const float* __restrict__ w, const float* __restrict__ b)
{
  __shared__ float s1[256], s2[256];
  const int r = blockIdx.x;                // (b*128 + t)*64 + c
  const int t = threadIdx.x;
  const int c  = r & 63;
  const int tt = (r >> 6) & 127;
  const int bb = r >> 13;
  const size_t inrow = ((size_t)bb * 64 + c) * 128 + tt;
  float v = co[inrow * 256 + t];
  s1[t] = v; s2[t] = v * v;
  __syncthreads();
  for (int o = 128; o > 0; o >>= 1) {
    if (t < o) { s1[t] += s1[t + o]; s2[t] += s2[t + o]; }
    __syncthreads();
  }
  float m    = s1[0] * (1.0f / 256.0f);
  float var  = s2[0] * (1.0f / 256.0f) - m * m;
  float rinv = rsqrtf(var + 1e-5f);
  float ln   = (v - m) * rinv * w[t] + b[t];
  out[(size_t)r * 256 + t] = x2[(size_t)r * 256 + t] + 0.5f * ln;
}

// (b, D1, D2, 256) -> (b, D2, D1, 256)
__global__ void k_transpose12(const float* __restrict__ in, float* __restrict__ out,
                              int D1, int D2)
{
  size_t id = (size_t)blockIdx.x * 256 + threadIdx.x;
  int d = id & 255;
  size_t rest = id >> 8;
  int j = rest % D2;
  size_t r2 = rest / D2;
  int i  = r2 % D1;
  int bb = r2 / D1;
  out[(((size_t)bb * D2 + j) * D1 + i) * 256 + d] = in[id];
}

// x2 (b,t,c,d) -> xc (b*c, d, t)
__global__ void k_to_ndt(const float* __restrict__ in, float* __restrict__ out)
{
  size_t id = (size_t)blockIdx.x * 256 + threadIdx.x;
  int d = id & 255;
  size_t rest = id >> 8;
  int c  = rest & 63;
  int t  = (rest >> 6) & 127;
  int bb = rest >> 13;
  out[(((size_t)bb * 64 + c) * 256 + d) * 128 + t] = in[id];
}

// gated residual: out = y*sig(z) + r*(1-sig(z))
__global__ void k_gate(const float* __restrict__ y, const float* __restrict__ r,
                       const float* __restrict__ z, float* __restrict__ out)
{
  size_t id = (size_t)blockIdx.x * 256 + threadIdx.x;
  float g = sigmoid_f(z[id]);
  out[id] = y[id] * g + r[id] * (1.0f - g);
}

// one dilated conv branch + exact GELU. x: (512, 256, 128); w: (28, 256, k)
__global__ void k_conv(const float* __restrict__ x, const float* __restrict__ w,
                       const float* __restrict__ bias, float* __restrict__ ybuf,
                       int k, int dil, int branch)
{
  int id = blockIdx.x * 256 + threadIdx.x;     // 512*28*128 exact
  int t = id & 127;
  int o = (id >> 7) % 28;
  int n = id / (28 * 128);
  int pad = (k - 1) * dil / 2;
  float sum = bias[o];
  const float* xn = x + (size_t)n * 256 * 128;
  const float* wr = w + (size_t)o * 256 * k;
  for (int ic = 0; ic < 256; ++ic) {
    const float* xi = xn + ic * 128;
    const float* wi = wr + ic * k;
    for (int kk = 0; kk < k; ++kk) {
      int tt = t + kk * dil - pad;
      if (tt >= 0 && tt < 128) sum += xi[tt] * wi[kk];
    }
  }
  ybuf[(size_t)n * 252 * 128 + ((size_t)branch * 28 + o) * 128 + t] = gelu_f(sum);
}

// GroupNorm per (n, branch, group of 7 ch x 128 t); writes (n*128+t, 256) cat
__global__ __launch_bounds__(128) void k_gn(
    const float* __restrict__ ybuf, const float* __restrict__ w,
    const float* __restrict__ b, float* __restrict__ cat)
{
  __shared__ float s1[128], s2[128];
  const int n = blockIdx.x, br = blockIdx.y, g = blockIdx.z;
  const int tid = threadIdx.x;
  const float* base = ybuf + (size_t)n * 252 * 128 + ((size_t)br * 28 + g * 7) * 128;
  float a = 0.0f, sq = 0.0f;
  for (int idx = tid; idx < 896; idx += 128) { float v = base[idx]; a += v; sq += v * v; }
  s1[tid] = a; s2[tid] = sq;
  __syncthreads();
  for (int o = 64; o > 0; o >>= 1) {
    if (tid < o) { s1[tid] += s1[tid + o]; s2[tid] += s2[tid + o]; }
    __syncthreads();
  }
  float m    = s1[0] * (1.0f / 896.0f);
  float var  = s2[0] * (1.0f / 896.0f) - m * m;
  float rinv = rsqrtf(var + 1e-5f);
  for (int idx = tid; idx < 896; idx += 128) {
    int ch = idx >> 7, t = idx & 127;
    int col = br * 28 + g * 7 + ch;
    cat[((size_t)n * 128 + t) * 256 + col] = (base[idx] - m) * rinv * w[col] + b[col];
  }
}

// zero the 4 pad columns (252..255) of the cat buffer
__global__ void k_zeropad(float* __restrict__ cat)
{
  int id = blockIdx.x * 256 + threadIdx.x;   // 65536*4 exact
  int r = id >> 2, c = 252 + (id & 3);
  cat[(size_t)r * 256 + c] = 0.0f;
}

// mean over heads of relpos -> (64,64) bias
__global__ void k_bias64(const float* __restrict__ relpos, float* __restrict__ bias64)
{
  int id = blockIdx.x * 256 + threadIdx.x;   // 4096 exact
  float s = 0.0f;
  for (int h = 0; h < 8; ++h) s += relpos[h * 4096 + id];
  bias64[id] = s * 0.125f;
}

// zero-pad proj_w (256,252) -> (256,256)
__global__ void k_padw(const float* __restrict__ pw, float* __restrict__ wpad)
{
  int id = blockIdx.x * 256 + threadIdx.x;   // 65536 exact
  int n = id >> 8, k = id & 255;
  wpad[id] = (k < 252) ? pw[n * 252 + k] : 0.0f;
}

// ---------------------------------------------------------------------------
// Host-side orchestration
// ---------------------------------------------------------------------------
static void launch_gemm(hipStream_t s, const float* A, int lda, const float* W, int ldw,
                        const float* bias, float* C, int M, int N, int K,
                        int act, bool accum)
{
  dim3 grid(N / 64, M / 64), blk(256);
  if (accum)         k_gemm<0, true ><<<grid, blk, 0, s>>>(A, lda, W, ldw, bias, C, M, N, K);
  else if (act == 1) k_gemm<1, false><<<grid, blk, 0, s>>>(A, lda, W, ldw, bias, C, M, N, K);
  else               k_gemm<0, false><<<grid, blk, 0, s>>>(A, lda, W, ldw, bias, C, M, N, K);
}

extern "C" void kernel_launch(void* const* d_in, const int* in_sizes, int n_in,
                              void* d_out, int out_size, void* d_ws, size_t ws_size,
                              hipStream_t stream)
{
  const float* x         = (const float*)d_in[0];
  const float* sa_wqkv   = (const float*)d_in[1];
  const float* sa_bqkv   = (const float*)d_in[2];
  const float* sa_wout   = (const float*)d_in[3];
  const float* sa_bout   = (const float*)d_in[4];
  const float* sa_relpos = (const float*)d_in[5];
  const float* sa_ln_w   = (const float*)d_in[6];
  const float* sa_ln_b   = (const float*)d_in[7];
  const float* ta_wqkv   = (const float*)d_in[8];
  const float* ta_bqkv   = (const float*)d_in[9];
  const float* ta_wout   = (const float*)d_in[10];
  const float* ta_bout   = (const float*)d_in[11];
  const float* ta_ln_w   = (const float*)d_in[12];
  const float* ta_ln_b   = (const float*)d_in[13];
  const float* cw[3]     = {(const float*)d_in[14], (const float*)d_in[15], (const float*)d_in[16]};
  const float* cb[3]     = {(const float*)d_in[17], (const float*)d_in[18], (const float*)d_in[19]};
  const float* gn_w      = (const float*)d_in[20];
  const float* gn_b      = (const float*)d_in[21];
  const float* proj_w    = (const float*)d_in[22];
  const float* proj_b    = (const float*)d_in[23];
  const float* conv_ln_w = (const float*)d_in[24];
  const float* conv_ln_b = (const float*)d_in[25];
  const float* ff_w1     = (const float*)d_in[26];
  const float* ff_b1     = (const float*)d_in[27];
  const float* ff_w2     = (const float*)d_in[28];
  const float* ff_b2     = (const float*)d_in[29];
  const float* ff_ln_w   = (const float*)d_in[30];
  const float* ff_ln_b   = (const float*)d_in[31];
  const float* g1_w      = (const float*)d_in[32];
  const float* g1_b      = (const float*)d_in[33];
  const float* g2_w      = (const float*)d_in[34];
  const float* g2_b      = (const float*)d_in[35];
  const float* g3_w      = (const float*)d_in[36];
  const float* g3_b      = (const float*)d_in[37];
  float* out = (float*)d_out;

  const int M  = 65536;                 // B*T*C
  const size_t NN = (size_t)M * 256;

  float* ws     = (float*)d_ws;
  float* bias64 = ws;                               // 4096
  float* wpad   = ws + 4096;                        // 65536
  float* bufA   = ws + 4096 + 65536;                // 65536*1024 f32 (qkv(bf16) / ybuf / h1)
  float* bufB   = bufA + (size_t)M * 1024;
  float* bufC   = bufB + NN;
  float* bufD   = bufC + NN;
  float* bufE   = bufD + NN;
  float* bufF   = bufE + NN;
  __bf16* qkvbuf = (__bf16*)bufA;                   // 65536*768 bf16 (96MB, fits in bufA)

  const int EB = 65536;   // blocks for NN-sized elementwise kernels (x256)

  // ---- prep ----
  k_bias64<<<16, 256, 0, stream>>>(sa_relpos, bias64);
  k_padw<<<256, 256, 0, stream>>>(proj_w, wpad);

  // ---- spatial attention ----
  k_gemm_obf16<<<dim3(12, 1024), 256, 0, stream>>>(x, 256, sa_wqkv, 256, sa_bqkv,
                                                   qkvbuf, M, 768, 256);
  k_attn<64, true><<<dim3(1024, 8), 128, 0, stream>>>(qkvbuf, bias64, bufB);
  launch_gemm(stream, bufB, 256, sa_wout, 256, sa_bout, bufF, M, 256, 256, 0, false);
  k_ln<<<M, 256, 0, stream>>>(bufF, x, bufE, sa_ln_w, sa_ln_b);              // so
  launch_gemm(stream, bufE, 256, g1_w,       512, g1_b,    bufB, M, 256, 256, 0, false);
  launch_gemm(stream, x,    256, g1_w + 256, 512, nullptr, bufB, M, 256, 256, 0, true);
  k_gate<<<EB, 256, 0, stream>>>(bufE, x, bufB, bufC);                       // x1

  // ---- temporal attention ----
  k_transpose12<<<EB, 256, 0, stream>>>(bufC, bufE, 128, 64);                // xt (b,c,t)
  k_gemm_obf16<<<dim3(12, 1024), 256, 0, stream>>>(bufE, 256, ta_wqkv, 256, ta_bqkv,
                                                   qkvbuf, M, 768, 256);
  k_attn<128, false><<<dim3(512, 8), 256, 0, stream>>>(qkvbuf, nullptr, bufB);
  launch_gemm(stream, bufB, 256, ta_wout, 256, ta_bout, bufF, M, 256, 256, 0, false);
  k_ln<<<M, 256, 0, stream>>>(bufF, bufE, bufB, ta_ln_w, ta_ln_b);           // to (b,c,t)
  k_transpose12<<<EB, 256, 0, stream>>>(bufB, bufE, 64, 128);                // to (b,t,c)
  launch_gemm(stream, bufE, 256, g2_w,       512, g2_b,    bufB, M, 256, 256, 0, false);
  launch_gemm(stream, bufC, 256, g2_w + 256, 512, nullptr, bufB, M, 256, 256, 0, true);
  k_gate<<<EB, 256, 0, stream>>>(bufE, bufC, bufB, bufD);                    // x2

  // ---- multi-scale dilated convs ----
  k_to_ndt<<<EB, 256, 0, stream>>>(bufD, bufE);                              // xc (n,d,t)
  {
    const int ksz[3]  = {3, 5, 7};
    const int dils[3] = {1, 2, 4};
    int br = 0;
    for (int ki = 0; ki < 3; ++ki)
      for (int di = 0; di < 3; ++di, ++br) {
        int k = ksz[ki], dil = dils[di];
        k_conv<<<7168, 256, 0, stream>>>(bufE, cw[ki] + (size_t)di * 28 * 256 * k,
                                         cb[ki] + di * 28, bufA, k, dil, br);
      }
  }
  k_gn<<<dim3(512, 9, 4), 128, 0, stream>>>(bufA, gn_w, gn_b, bufF);         // cat
  k_zeropad<<<1024, 256, 0, stream>>>(bufF);
  launch_gemm(stream, bufF, 256, wpad, 256, proj_b, bufB, M, 256, 256, 0, false);  // co (b,c,t)
  k_conv_ln_res<<<M, 256, 0, stream>>>(bufB, bufD, bufC, conv_ln_w, conv_ln_b);    // x3

  // ---- FFN + final gate ----
  launch_gemm(stream, bufC, 256,  ff_w1, 256,  ff_b1, bufA, M, 1024, 256, 1, false);
  launch_gemm(stream, bufA, 1024, ff_w2, 1024, ff_b2, bufE, M, 256, 1024, 0, false);
  k_ln<<<M, 256, 0, stream>>>(bufE, nullptr, bufF, ff_ln_w, ff_ln_b);        // fo
  launch_gemm(stream, bufF, 256, g3_w,       512, g3_b,    bufB, M, 256, 256, 0, false);
  launch_gemm(stream, bufC, 256, g3_w + 256, 512, nullptr, bufB, M, 256, 256, 0, true);
  k_gate<<<EB, 256, 0, stream>>>(bufF, bufC, bufB, out);
}
